// GraphGPSRegressor_11553462026955
// MI455X (gfx1250) — matrix operand order repaired and checked by reading, not compiled, not run on hardware
//
#include <hip/hip_runtime.h>
#include <math.h>

#define Nn   512
#define Hd   128
#define Ed   32768
#define NH   4
#define DH   32
#define PR   64

typedef __bf16 bf16_t;
typedef __attribute__((ext_vector_type(16))) bf16_t v16bf;
typedef __attribute__((ext_vector_type(4)))  bf16_t bf16x4;
typedef __attribute__((ext_vector_type(2)))  bf16_t bf16x2;
typedef __attribute__((ext_vector_type(8)))  float  v8f;

#define WMMA_BF16(a,b,c) __builtin_amdgcn_wmma_f32_16x16x32_bf16(false,(a),false,(b),(short)0,(c),false,false)

__device__ __forceinline__ float gelu_f(float x){
  return 0.5f * x * (1.0f + erff(x * 0.70710678118654752440f));
}
__device__ __forceinline__ v8f vzero8(){
  v8f z;
  #pragma unroll
  for (int i=0;i<8;i++) z[i]=0.0f;
  return z;
}
__device__ __forceinline__ bf16x4 pack4(float a, float b, float c, float d){
  bf16x4 p; p[0]=(bf16_t)a; p[1]=(bf16_t)b; p[2]=(bf16_t)c; p[3]=(bf16_t)d;
  return p;
}
// A-matrix 16x32 bf16 fragment per ISA table: lane r=lane&15, kbase=8*(lane>>4),
// elements 0..7 -> K=kbase..kbase+7 ; elements 8..15 -> K=kbase+16..kbase+23
__device__ __forceinline__ v16bf frag_a(const bf16_t* base, int stride, int lane){
  int r  = lane & 15;
  int kb = (lane >> 4) << 3;
  const bf16_t* p = base + r * stride + kb;
  v16bf f;
  #pragma unroll
  for (int e=0;e<8;e++){ f[e] = p[e]; f[e+8] = p[e+16]; }
  return f;
}
// B-matrix 32x16 bf16 fragment: buffer stored [n][k]; kbase=8*(lane>>4),
// elements 0..7 -> K=kbase..kbase+7 ; elements 8..15 -> K=kbase+16..kbase+23, n = lane&15
__device__ __forceinline__ v16bf frag_b(const bf16_t* base, int stride, int lane){
  int n  = lane & 15;
  int kb = (lane >> 4) << 3;
  const bf16_t* p = base + n * stride + kb;
  v16bf f;
  #pragma unroll
  for (int e=0;e<8;e++){ f[e] = p[e]; f[e+8] = p[e+16]; }
  return f;
}
// Same but buffer stored [k][n] (k-major), e.g. V tiles
__device__ __forceinline__ v16bf frag_b_kmaj(const bf16_t* base, int stride, int lane){
  int n  = lane & 15;
  int kb = (lane >> 4) << 3;
  v16bf f;
  #pragma unroll
  for (int e=0;e<8;e++){ f[e] = base[(kb + e) * stride + n]; f[e+8] = base[(kb + 16 + e) * stride + n]; }
  return f;
}

// ---------------------------------------------------------------------------
// Generic WMMA GEMM: C[M,N] = act(A[M,K] @ W[N,K]^T + bias) (+ resid)
// Requires M%128==0, N%64==0, K%32==0 (true for every call site).
// grid (M/128, N/64), block 256 (8 waves).
// ---------------------------------------------------------------------------
#define LDT 36
__global__ __launch_bounds__(256) void gemm_wmma_kernel(
    const float* __restrict__ A, const float* __restrict__ W,
    const float* __restrict__ bias, const float* __restrict__ resid,
    float* __restrict__ C, int M, int K, int N, int act)
{
  __shared__ bf16_t Ab[128][LDT];
  __shared__ bf16_t Bb[64][LDT];
  const int tid  = threadIdx.x;
  const int lane = tid & 31;
  const int wv   = tid >> 5;
  const int m0   = blockIdx.x * 128;
  const int n0   = blockIdx.y * 64;

  v8f acc[4];
  #pragma unroll
  for (int ns=0; ns<4; ns++) acc[ns] = vzero8();

  for (int kc = 0; kc < K; kc += 32){
    if (lane == 0)
      __builtin_prefetch(&A[(size_t)(m0 + wv*16) * K + kc + 32], 0, 1);
    // A: 128x32 elems = 1024 quads / 256 thr = 4 each (float4 -> bf16x4)
    #pragma unroll
    for (int it=0; it<4; it++){
      int idx = tid + it*256; int r = idx >> 3, c = (idx & 7) * 4;
      float4 v = *(const float4*)(A + (size_t)(m0 + r) * K + kc + c);
      *(bf16x4*)&Ab[r][c] = pack4(v.x, v.y, v.z, v.w);
    }
    // B: 64x32 = 512 quads / 256 thr = 2 each
    #pragma unroll
    for (int it=0; it<2; it++){
      int idx = tid + it*256; int r = idx >> 3, c = (idx & 7) * 4;
      float4 v = *(const float4*)(W + (size_t)(n0 + r) * K + kc + c);
      *(bf16x4*)&Bb[r][c] = pack4(v.x, v.y, v.z, v.w);
    }
    __syncthreads();
    v16bf a = frag_a(&Ab[wv*16][0], LDT, lane);
    #pragma unroll
    for (int ns=0; ns<4; ns++){
      v16bf b = frag_b(&Bb[ns*16][0], LDT, lane);
      acc[ns] = WMMA_BF16(a, b, acc[ns]);
    }
    __syncthreads();
  }

  const int half = lane >> 4, nl = lane & 15;
  #pragma unroll
  for (int ns=0; ns<4; ns++){
    #pragma unroll
    for (int r=0; r<8; r++){
      int m = m0 + wv*16 + r + 8*half;
      int n = n0 + ns*16 + nl;
      float v = acc[ns][r];
      if (bias)  v += bias[n];
      if (act)   v  = gelu_f(v);
      if (resid) v += resid[(size_t)m * N + n];
      C[(size_t)m * N + n] = v;
    }
  }
}

// ---------------------------------------------------------------------------
// WMMA attention: one wave per (16-query tile, head). K/V staged bf16 in LDS.
// ---------------------------------------------------------------------------
__global__ __launch_bounds__(32) void attn_kernel(
    const float* __restrict__ qkv, float* __restrict__ out)
{
  __shared__ bf16_t Kb[Nn][LDT];
  __shared__ bf16_t Vb[Nn][LDT];
  __shared__ float  Sb[16][Nn];
  __shared__ bf16_t Pb[16][Nn + 32];
  const int lane = threadIdx.x;
  const int m0   = blockIdx.x * 16;
  const int h    = blockIdx.y;
  const float scale = 0.17677669529663688f; // 1/sqrt(32)

  // stage K/V: 512x32 elems = 4096 quads each, 128 per lane
  for (int idx = lane; idx < (Nn * DH) / 4; idx += 32){
    int r = idx >> 3, c = (idx & 7) * 4;
    float4 kv = *(const float4*)(qkv + (size_t)r * 384 + 128 + h*DH + c);
    float4 vv = *(const float4*)(qkv + (size_t)r * 384 + 256 + h*DH + c);
    *(bf16x4*)&Kb[r][c] = pack4(kv.x, kv.y, kv.z, kv.w);
    *(bf16x4*)&Vb[r][c] = pack4(vv.x, vv.y, vv.z, vv.w);
  }
  // Q fragment straight from global
  v16bf qa;
  {
    int r  = lane & 15;
    int kb = (lane >> 4) << 3;
    const float* qp = qkv + (size_t)(m0 + r) * 384 + h*DH;
    #pragma unroll
    for (int e=0;e<8;e++){ qa[e] = (bf16_t)qp[kb+e]; qa[e+8] = (bf16_t)qp[kb+16+e]; }
  }
  __syncthreads();

  const int half = lane >> 4, nl = lane & 15;
  for (int kt = 0; kt < Nn/16; kt++){
    v16bf b = frag_b(&Kb[kt*16][0], LDT, lane);
    v8f s = vzero8();
    s = WMMA_BF16(qa, b, s);
    #pragma unroll
    for (int r=0; r<8; r++)
      Sb[r + 8*half][kt*16 + nl] = s[r] * scale;
  }
  __syncthreads();

  if (lane < 16){
    int row = lane;
    float mx = -3.4e38f;
    for (int j=0;j<Nn;j++) mx = fmaxf(mx, Sb[row][j]);
    float sum = 0.0f;
    for (int j=0;j<Nn;j++){ float e = __expf(Sb[row][j] - mx); Sb[row][j] = e; sum += e; }
    float inv = 1.0f / sum;
    for (int j=0;j<Nn;j+=4)
      *(bf16x4*)&Pb[row][j] = pack4(Sb[row][j]*inv, Sb[row][j+1]*inv,
                                    Sb[row][j+2]*inv, Sb[row][j+3]*inv);
  }
  __syncthreads();

  v8f o0 = vzero8(), o1 = vzero8();
  for (int kc = 0; kc < Nn/32; kc++){
    v16bf a  = frag_a(&Pb[0][kc*32], Nn + 32, lane);
    v16bf b0 = frag_b_kmaj(&Vb[kc*32][0],  LDT, lane);
    v16bf b1 = frag_b_kmaj(&Vb[kc*32][16], LDT, lane);
    o0 = WMMA_BF16(a, b0, o0);
    o1 = WMMA_BF16(a, b1, o1);
  }
  #pragma unroll
  for (int r=0; r<8; r++){
    int m = m0 + r + 8*half;
    out[(size_t)m * Hd + h*DH +      nl] = o0[r];
    out[(size_t)m * Hd + h*DH + 16 + nl] = o1[r];
  }
}

// ---------------------------------------------------------------------------
// Fused pair head: block = (i, 128 j's). mapper l1 (K=516 pad 544) -> GELU ->
// mapper l2 -> cond write -> dec_ln -> dec1 -> GELU -> dec2 dot -> pred
// Segment selection for pf chunks is uniform per K-chunk (no divergence).
// ---------------------------------------------------------------------------
#define LDX 132
#define LDL 68
__global__ __launch_bounds__(256) void pair_head_kernel(
    const float* __restrict__ x, const float* __restrict__ geo,
    const float* __restrict__ W1, const float* __restrict__ b1,
    const float* __restrict__ W2, const float* __restrict__ b2,
    const float* __restrict__ lng, const float* __restrict__ lnb,
    const float* __restrict__ Wd1, const float* __restrict__ bd1,
    const float* __restrict__ wd2, const float* __restrict__ bd2,
    float* __restrict__ cond, float* __restrict__ pred)
{
  __shared__ float  xi[Hd];
  __shared__ float  wd2s[Hd];
  __shared__ bf16_t xj[128][LDX];
  __shared__ float  geos[128][4];
  __shared__ bf16_t Ab[128][LDT];
  __shared__ bf16_t Bb[128][LDT];
  __shared__ bf16_t h1[128][LDX];
  __shared__ float  lat[128][LDL];
  __shared__ bf16_t latn[128][LDL];

  const int i   = blockIdx.x;
  const int j0  = blockIdx.y * 128;
  const int tid = threadIdx.x;
  const int lane = tid & 31, wv = tid >> 5;
  const int half = lane >> 4, nl = lane & 15;

  if (tid < Hd){ xi[tid] = x[(size_t)i * Hd + tid]; wd2s[tid] = wd2[tid]; }
  // xj: 128x128 = 4096 quads / 256 thr = 16 each
  #pragma unroll
  for (int it=0; it<16; it++){
    int idx = tid + it*256; int r = idx >> 5, c = (idx & 31) * 4;
    float4 v = *(const float4*)(x + (size_t)(j0 + r) * Hd + c);
    *(bf16x4*)&xj[r][c] = pack4(v.x, v.y, v.z, v.w);
  }
  if (tid < 128){
    float4 g = *(const float4*)(geo + ((size_t)i * Nn + j0 + tid) * 4);
    geos[tid][0]=g.x; geos[tid][1]=g.y; geos[tid][2]=g.z; geos[tid][3]=g.w;
  }
  __syncthreads();

  // ---- mapper l1: [128,544pad] @ W1[128,516]^T ----
  v8f acc[8];
  #pragma unroll
  for (int ns=0; ns<8; ns++) acc[ns] = vzero8();

  for (int kc = 0; kc < 544; kc += 32){
    // build A chunk (uniform segment per chunk): 1024 quads / 256 thr = 4 each
    if (kc < 128){
      #pragma unroll
      for (int it=0; it<4; it++){
        int idx = tid + it*256; int r = idx >> 3, c = (idx & 7) * 4;
        float4 v = *(const float4*)(xi + kc + c);
        *(bf16x4*)&Ab[r][c] = pack4(v.x, v.y, v.z, v.w);
      }
    } else if (kc < 256){
      #pragma unroll
      for (int it=0; it<4; it++){
        int idx = tid + it*256; int r = idx >> 3, c = (idx & 7) * 4;
        *(bf16x4*)&Ab[r][c] = *(const bf16x4*)&xj[r][kc - 128 + c];
      }
    } else if (kc < 384){
      #pragma unroll
      for (int it=0; it<4; it++){
        int idx = tid + it*256; int r = idx >> 3, c = (idx & 7) * 4;
        float4 a = *(const float4*)(xi + kc - 256 + c);
        bf16x4 bx = *(const bf16x4*)&xj[r][kc - 256 + c];
        *(bf16x4*)&Ab[r][c] = pack4(a.x*(float)bx[0], a.y*(float)bx[1],
                                    a.z*(float)bx[2], a.w*(float)bx[3]);
      }
    } else if (kc < 512){
      #pragma unroll
      for (int it=0; it<4; it++){
        int idx = tid + it*256; int r = idx >> 3, c = (idx & 7) * 4;
        float4 a = *(const float4*)(xi + kc - 384 + c);
        bf16x4 bx = *(const bf16x4*)&xj[r][kc - 384 + c];
        *(bf16x4*)&Ab[r][c] = pack4(a.x-(float)bx[0], a.y-(float)bx[1],
                                    a.z-(float)bx[2], a.w-(float)bx[3]);
      }
    } else { // geo tail: k = 512..515 is exactly one quad, rest zero
      #pragma unroll
      for (int it=0; it<4; it++){
        int idx = tid + it*256; int r = idx >> 3, c = (idx & 7) * 4;
        if (c == 0)
          *(bf16x4*)&Ab[r][c] = pack4(geos[r][0], geos[r][1], geos[r][2], geos[r][3]);
        else
          *(bf16x4*)&Ab[r][c] = pack4(0.f, 0.f, 0.f, 0.f);
      }
    }
    // B chunk from W1 (row stride 516 floats; 516*4 bytes % 16 == 0)
    if (kc < 512){
      #pragma unroll
      for (int it=0; it<4; it++){
        int idx = tid + it*256; int r = idx >> 3, c = (idx & 7) * 4;
        float4 v = *(const float4*)(W1 + (size_t)r * 516 + kc + c);
        *(bf16x4*)&Bb[r][c] = pack4(v.x, v.y, v.z, v.w);
      }
    } else {
      #pragma unroll
      for (int it=0; it<4; it++){
        int idx = tid + it*256; int r = idx >> 3, c = (idx & 7) * 4;
        if (c == 0){
          float4 v = *(const float4*)(W1 + (size_t)r * 516 + 512);
          *(bf16x4*)&Bb[r][c] = pack4(v.x, v.y, v.z, v.w);
        } else
          *(bf16x4*)&Bb[r][c] = pack4(0.f, 0.f, 0.f, 0.f);
      }
    }
    __syncthreads();
    v16bf a = frag_a(&Ab[wv*16][0], LDT, lane);
    #pragma unroll
    for (int ns=0; ns<8; ns++){
      v16bf b = frag_b(&Bb[ns*16][0], LDT, lane);
      acc[ns] = WMMA_BF16(a, b, acc[ns]);
    }
    __syncthreads();
  }
  #pragma unroll
  for (int ns=0; ns<8; ns++)
    #pragma unroll
    for (int r=0; r<8; r++){
      int m = wv*16 + r + 8*half, n = ns*16 + nl;
      h1[m][n] = (bf16_t)gelu_f(acc[ns][r] + b1[n]);
    }
  __syncthreads();

  // ---- mapper l2: [128,128] @ W2[64,128]^T ----
  v8f acc2[4];
  #pragma unroll
  for (int ns=0; ns<4; ns++) acc2[ns] = vzero8();
  for (int kc = 0; kc < 128; kc += 32){
    #pragma unroll
    for (int it=0; it<2; it++){
      int idx = tid + it*256; int r = idx >> 3, c = (idx & 7) * 4;
      float4 v = *(const float4*)(W2 + (size_t)r * 128 + kc + c);
      *(bf16x4*)&Bb[r][c] = pack4(v.x, v.y, v.z, v.w);
    }
    __syncthreads();
    v16bf a = frag_a(&h1[wv*16][kc], LDX, lane);
    #pragma unroll
    for (int ns=0; ns<4; ns++){
      v16bf b = frag_b(&Bb[ns*16][0], LDT, lane);
      acc2[ns] = WMMA_BF16(a, b, acc2[ns]);
    }
    __syncthreads();
  }
  #pragma unroll
  for (int ns=0; ns<4; ns++)
    #pragma unroll
    for (int r=0; r<8; r++){
      int m = wv*16 + r + 8*half, n = ns*16 + nl;
      lat[m][n] = acc2[ns][r] + b2[n];
    }
  __syncthreads();

  // cond[p][i][j] = lat[j][p]
  #pragma unroll
  for (int it=0; it<32; it++){
    int idx = it*256 + tid; int j = idx & 127, p = idx >> 7;
    cond[(size_t)p * (Nn*Nn) + (size_t)i * Nn + j0 + j] = lat[j][p];
  }

  // dec_ln over 64 per row
  if (tid < 128){
    int row = tid;
    float m = 0.0f;
    for (int n=0;n<PR;n++) m += lat[row][n];
    m *= (1.0f/PR);
    float v = 0.0f;
    for (int n=0;n<PR;n++){ float d = lat[row][n]-m; v += d*d; }
    float rinv = rsqrtf(v*(1.0f/PR) + 1e-5f);
    for (int n=0;n<PR;n+=2){
      bf16x2 p;
      p[0] = (bf16_t)((lat[row][n  ]-m)*rinv*lng[n  ] + lnb[n  ]);
      p[1] = (bf16_t)((lat[row][n+1]-m)*rinv*lng[n+1] + lnb[n+1]);
      *(bf16x2*)&latn[row][n] = p;
    }
  }
  __syncthreads();

  // dec1: [128,64] @ Wd1[128,64]^T -> GELU -> h1 (reuse)
  v8f accd[8];
  #pragma unroll
  for (int ns=0; ns<8; ns++) accd[ns] = vzero8();
  for (int kc = 0; kc < 64; kc += 32){
    #pragma unroll
    for (int it=0; it<4; it++){
      int idx = tid + it*256; int r = idx >> 3, c = (idx & 7) * 4;
      float4 v = *(const float4*)(Wd1 + (size_t)r * 64 + kc + c);
      *(bf16x4*)&Bb[r][c] = pack4(v.x, v.y, v.z, v.w);
    }
    __syncthreads();
    v16bf a = frag_a(&latn[wv*16][kc], LDL, lane);
    #pragma unroll
    for (int ns=0; ns<8; ns++){
      v16bf b = frag_b(&Bb[ns*16][0], LDT, lane);
      accd[ns] = WMMA_BF16(a, b, accd[ns]);
    }
    __syncthreads();
  }
  #pragma unroll
  for (int ns=0; ns<8; ns++)
    #pragma unroll
    for (int r=0; r<8; r++){
      int m = wv*16 + r + 8*half, n = ns*16 + nl;
      h1[m][n] = (bf16_t)gelu_f(accd[ns][r] + bd1[n]);
    }
  __syncthreads();

  if (tid < 128){
    int row = tid;
    float s = bd2[0];
    for (int c=0;c<Hd;c++) s += (float)h1[row][c] * wd2s[c];
    pred[(size_t)i * Nn + j0 + row] = s;
  }
}

// ---------------------------------------------------------------------------
// small VALU kernels
// ---------------------------------------------------------------------------
__global__ __launch_bounds__(128) void node_enc_l1_kernel(
    const float* __restrict__ xn, const float* __restrict__ se,
    const float* __restrict__ lap, const float* __restrict__ W,
    const float* __restrict__ b, float* __restrict__ h)
{
  __shared__ float in[13];
  int node = blockIdx.x, t = threadIdx.x;
  if      (t < 3)  in[t] = xn[node*3 + t];
  else if (t < 5)  in[t] = se[node*2 + (t-3)];
  else if (t < 13) in[t] = lap[node*8 + (t-5)];
  __syncthreads();
  float s = b[t];
  #pragma unroll
  for (int k=0;k<13;k++) s += W[t*13 + k] * in[k];
  h[(size_t)node * Hd + t] = gelu_f(s);
}

__global__ __launch_bounds__(128) void edge_enc_l1_kernel(
    const float* __restrict__ ea, const float* __restrict__ W,
    const float* __restrict__ b, float* __restrict__ h)
{
  int e = blockIdx.x, t = threadIdx.x;
  float a0 = ea[e*3+0], a1 = ea[e*3+1], a2 = ea[e*3+2];
  float s = b[t] + W[t*3+0]*a0 + W[t*3+1]*a1 + W[t*3+2]*a2;
  h[(size_t)e * Hd + t] = gelu_f(s);
}

__global__ __launch_bounds__(128) void ln_kernel(
    const float* __restrict__ x, const float* __restrict__ g,
    const float* __restrict__ b, float* __restrict__ y)
{
  __shared__ float red[128];
  int row = blockIdx.x, t = threadIdx.x;
  float v = x[(size_t)row * Hd + t];
  red[t] = v; __syncthreads();
  #pragma unroll
  for (int s=64; s>0; s>>=1){ if (t < s) red[t] += red[t+s]; __syncthreads(); }
  float mean = red[0] * (1.0f/Hd); __syncthreads();
  float d = v - mean;
  red[t] = d*d; __syncthreads();
  #pragma unroll
  for (int s=64; s>0; s>>=1){ if (t < s) red[t] += red[t+s]; __syncthreads(); }
  float rinv = rsqrtf(red[0] * (1.0f/Hd) + 1e-5f);
  y[(size_t)row * Hd + t] = d * rinv * g[t] + b[t];
}

__global__ __launch_bounds__(256) void gather_add_kernel(
    const float* __restrict__ xn, const float* __restrict__ ee,
    const int* __restrict__ src, float* __restrict__ out)
{
  size_t idx = (size_t)blockIdx.x * 256 + threadIdx.x;
  int e = (int)(idx >> 7), c = (int)(idx & 127);
  out[idx] = xn[(size_t)src[e] * Hd + c] + ee[idx];
}

__global__ __launch_bounds__(256) void scatter_add_kernel(
    const float* __restrict__ msg, const int* __restrict__ dst,
    float* __restrict__ agg)
{
  size_t idx = (size_t)blockIdx.x * 256 + threadIdx.x;
  int e = (int)(idx >> 7), c = (int)(idx & 127);
  atomicAdd(&agg[(size_t)dst[e] * Hd + c], msg[idx]);
}

__global__ __launch_bounds__(256) void upd_in_kernel(
    const float* __restrict__ xn, const float* __restrict__ agg,
    const float* __restrict__ eps, float* __restrict__ u)
{
  size_t idx = (size_t)blockIdx.x * 256 + threadIdx.x;
  u[idx] = (1.0f + eps[0]) * xn[idx] + agg[idx];
}

// ---------------------------------------------------------------------------
extern "C" void kernel_launch(void* const* d_in, const int* in_sizes, int n_in,
                              void* d_out, int out_size, void* d_ws, size_t ws_size,
                              hipStream_t stream)
{
  (void)in_sizes; (void)n_in; (void)out_size; (void)ws_size;
  const float* x_node = (const float*)d_in[0];
  const float* se_f   = (const float*)d_in[1];
  const float* lap    = (const float*)d_in[2];
  const float* e_attr = (const float*)d_in[3];
  const float* p_geo  = (const float*)d_in[4];
  const int*   eidx   = (const int*)d_in[5];

  // params flattened by sorted dict keys (depth-first)
  int pi = 6;
  auto P = [&](void)->const float* { return (const float*)d_in[pi++]; };
  const float *dec1W = P(), *dec1b = P();
  const float *dec2W = P(), *dec2b = P();
  const float *declnb = P(), *declng = P();
  const float *ee1W = P(), *ee1b = P(), *ee2W = P(), *ee2b = P();
  struct LayerP {
    const float *Wo,*Wqkv,*bo,*bqkv;
    const float *em1W,*em1b,*em2W,*em2b;
    const float *eps;
    const float *f1W,*f1b,*f2W,*f2b;
    const float *ln1b,*ln1g,*ln2b,*ln2g,*ln3b,*ln3g;
    const float *um1W,*um1b,*um2W,*um2b;
  } L[4];
  for (int l=0;l<4;l++){
    L[l].Wo=P(); L[l].Wqkv=P(); L[l].bo=P(); L[l].bqkv=P();
    L[l].em1W=P(); L[l].em1b=P(); L[l].em2W=P(); L[l].em2b=P();
    L[l].eps=P();
    L[l].f1W=P(); L[l].f1b=P(); L[l].f2W=P(); L[l].f2b=P();
    L[l].ln1b=P(); L[l].ln1g=P(); L[l].ln2b=P(); L[l].ln2g=P();
    L[l].ln3b=P(); L[l].ln3g=P();
    L[l].um1W=P(); L[l].um1b=P(); L[l].um2W=P(); L[l].um2b=P();
  }
  const float *mp1W = P(), *mp1b = P(), *mp2W = P(), *mp2b = P();
  const float *ne1W = P(), *ne1b = P(), *ne2W = P(), *ne2b = P();

  // workspace bump allocator
  char* wsp = (char*)d_ws; size_t off = 0;
  auto alloc = [&](size_t bytes)->float* {
    float* p = (float*)(wsp + off);
    off += (bytes + 255) & ~(size_t)255;
    return p;
  };
  float* xbuf = alloc((size_t)Nn*Hd*4);
  float* xn   = alloc((size_t)Nn*Hd*4);
  float* agg  = alloc((size_t)Nn*Hd*4);
  float* ubuf = alloc((size_t)Nn*Hd*4);
  float* xg   = alloc((size_t)Nn*Hd*4);
  float* qkv  = alloc((size_t)Nn*384*4);
  float* attn = alloc((size_t)Nn*Hd*4);
  float* hwide= alloc((size_t)Nn*512*4);        // FFN / upd / node-enc hidden
  float* eemb = alloc((size_t)Ed*Hd*4);
  float* ebA  = alloc((size_t)Ed*Hd*4);
  float* ebB  = alloc((size_t)Ed*Hd*4);

  float* out   = (float*)d_out;
  float* out_x = out;
  float* cond  = out + (size_t)Nn*Hd;
  float* pred  = cond + (size_t)PR*Nn*Nn;

  const int* src = eidx;
  const int* dst = eidx + Ed;

  // ---- encoders ----
  node_enc_l1_kernel<<<Nn, 128, 0, stream>>>(x_node, se_f, lap, ne1W, ne1b, hwide);
  gemm_wmma_kernel<<<dim3(Nn/128, Hd/64), 256, 0, stream>>>(
      hwide, ne2W, ne2b, nullptr, xbuf, Nn, Hd, Hd, 0);
  edge_enc_l1_kernel<<<Ed, 128, 0, stream>>>(e_attr, ee1W, ee1b, ebA);
  gemm_wmma_kernel<<<dim3(Ed/128, Hd/64), 256, 0, stream>>>(
      ebA, ee2W, ee2b, nullptr, eemb, Ed, Hd, Hd, 0);

  // ---- layers ----
  for (int l=0;l<4;l++){
    // GINE conv
    ln_kernel<<<Nn, 128, 0, stream>>>(xbuf, L[l].ln1g, L[l].ln1b, xn);
    gather_add_kernel<<<(Ed*Hd)/256, 256, 0, stream>>>(xn, eemb, src, ebA);
    gemm_wmma_kernel<<<dim3(Ed/128, Hd/64), 256, 0, stream>>>(
        ebA, L[l].em1W, L[l].em1b, nullptr, ebB, Ed, Hd, Hd, 1);
    gemm_wmma_kernel<<<dim3(Ed/128, Hd/64), 256, 0, stream>>>(
        ebB, L[l].em2W, L[l].em2b, nullptr, ebA, Ed, Hd, Hd, 0);
    hipMemsetAsync(agg, 0, (size_t)Nn*Hd*4, stream);
    scatter_add_kernel<<<(Ed*Hd)/256, 256, 0, stream>>>(ebA, dst, agg);
    upd_in_kernel<<<(Nn*Hd)/256, 256, 0, stream>>>(xn, agg, L[l].eps, ubuf);
    gemm_wmma_kernel<<<dim3(Nn/128, Hd/64), 256, 0, stream>>>(
        ubuf, L[l].um1W, L[l].um1b, nullptr, hwide, Nn, Hd, Hd, 1);
    gemm_wmma_kernel<<<dim3(Nn/128, Hd/64), 256, 0, stream>>>(
        hwide, L[l].um2W, L[l].um2b, xbuf, xbuf, Nn, Hd, Hd, 0);
    // attention
    ln_kernel<<<Nn, 128, 0, stream>>>(xbuf, L[l].ln2g, L[l].ln2b, xg);
    gemm_wmma_kernel<<<dim3(Nn/128, 384/64), 256, 0, stream>>>(
        xg, L[l].Wqkv, L[l].bqkv, nullptr, qkv, Nn, Hd, 384, 0);
    attn_kernel<<<dim3(Nn/16, NH), 32, 0, stream>>>(qkv, attn);
    gemm_wmma_kernel<<<dim3(Nn/128, Hd/64), 256, 0, stream>>>(
        attn, L[l].Wo, L[l].bo, xbuf, xbuf, Nn, Hd, Hd, 0);
    // FFN
    ln_kernel<<<Nn, 128, 0, stream>>>(xbuf, L[l].ln3g, L[l].ln3b, xg);
    gemm_wmma_kernel<<<dim3(Nn/128, 512/64), 256, 0, stream>>>(
        xg, L[l].f1W, L[l].f1b, nullptr, hwide, Nn, Hd, 512, 1);
    gemm_wmma_kernel<<<dim3(Nn/128, Hd/64), 256, 0, stream>>>(
        hwide, L[l].f2W, L[l].f2b, xbuf, xbuf, Nn, 512, Hd, 0);
  }

  // ---- outputs ----
  hipMemcpyAsync(out_x, xbuf, (size_t)Nn*Hd*4, hipMemcpyDeviceToDevice, stream);
  pair_head_kernel<<<dim3(Nn, Nn/128), 256, 0, stream>>>(
      xbuf, p_geo, mp1W, mp1b, mp2W, mp2b, declng, declnb,
      dec1W, dec1b, dec2W, dec2b, cond, pred);
}
